// NVFP4QDQModule_56453050139063
// MI455X (gfx1250) — compile-verified, matrix operand order checked by reading, not run on hardware
//
#include <hip/hip_runtime.h>

// ---------------------------------------------------------------------------
// NVFP4 QDQ linear for MI455X (gfx1250, wave32, WMMA).
// out = (s2*ws2) * (qi*qs) @ (w4*ws)^T + bias   -- exact in bf16 operands.
// ---------------------------------------------------------------------------

typedef __attribute__((ext_vector_type(16))) __bf16 v16bf;
typedef __attribute__((ext_vector_type(8)))  float  v8f;

union Frag32B {          // 32-byte fragment: 16 bf16 elements
    v16bf v;
    uint4 u[2];
    __bf16 h[16];
};

// ----- E2M1 quantize->dequantize, faithful to searchsorted(left) + tie-up at
// 0.75 / 1.75 / 3.5 (reference _cast_fp4 then E2M1_VALUES lookup). -----------
__device__ __forceinline__ float e2m1_qdq(float x) {
    float a = __builtin_fabsf(x);
    int ord = (a > 0.25f) + (a > 0.75f) + (a > 1.25f) + (a > 1.75f) +
              (a > 2.5f)  + (a > 3.5f)  + (a > 5.0f);
    ord += (int)((a == 0.75f) | (a == 1.75f) | (a == 3.5f));
    float v = (ord < 4)
        ? ((ord < 2) ? (ord == 0 ? 0.0f : 0.5f) : (ord == 2 ? 1.0f : 1.5f))
        : ((ord < 6) ? (ord == 4 ? 2.0f : 3.0f) : (ord == 6 ? 4.0f : 6.0f));
    return __builtin_copysignf(v, x);
}

// ----- round-trip through float8_e4m3fn (clamp to +-448, RNE to 3 mantissa
// bits, denormal quantum 2^-9). ----------------------------------------------
__device__ __forceinline__ float fp8_e4m3_round(float s) {
    float a = __builtin_fabsf(s);
    a = fminf(a, 448.0f);
    float r;
    if (a < 0.015625f) {                       // below 2^-6: e4m3 subnormals
        r = rintf(a * 512.0f) * 0.001953125f;  // quantum 2^-9, RNE
    } else {
        unsigned u = __float_as_uint(a);
        unsigned rounded = u + 0x7FFFFu + ((u >> 20) & 1u);  // RNE to 3 bits
        rounded &= 0xFFF00000u;
        r = fminf(__uint_as_float(rounded), 448.0f);
    }
    return __builtin_copysignf(r, s);
}

// ---------------------------------------------------------------------------
// Kernel 1: global amax(|x|) reduction -> atomicMax on float bits (x >= 0).
// ---------------------------------------------------------------------------
__global__ void __launch_bounds__(256)
amax_kernel(const float4* __restrict__ x4, size_t n4, unsigned* __restrict__ out) {
    __shared__ float red[256];
    size_t i = (size_t)blockIdx.x * blockDim.x + threadIdx.x;
    size_t stride = (size_t)gridDim.x * blockDim.x;
    float m = 0.0f;
    for (; i < n4; i += stride) {
        float4 v = x4[i];
        m = fmaxf(m, fmaxf(fmaxf(__builtin_fabsf(v.x), __builtin_fabsf(v.y)),
                           fmaxf(__builtin_fabsf(v.z), __builtin_fabsf(v.w))));
    }
    red[threadIdx.x] = m;
    __syncthreads();
    for (int s = 128; s > 0; s >>= 1) {
        if ((int)threadIdx.x < s)
            red[threadIdx.x] = fmaxf(red[threadIdx.x], red[threadIdx.x + s]);
        __syncthreads();
    }
    if (threadIdx.x == 0) atomicMax(out, __float_as_uint(red[0]));
}

// ---------------------------------------------------------------------------
// Kernel 2: weight QDQ -> bf16  Bq[o,k] = e2m1(w/(ws*ws2)) * ws   (exact bf16)
// One thread per 16-element block.
// ---------------------------------------------------------------------------
__global__ void __launch_bounds__(256)
wqdq_kernel(const float* __restrict__ w, const float* __restrict__ wscale,
            const float* __restrict__ wscale2, __bf16* __restrict__ Bq,
            int O, int K) {
    int nkb = K >> 4;
    int idx = blockIdx.x * 256 + threadIdx.x;       // (o, kb) flat
    if (idx >= O * nkb) return;
    float wsv = wscale[idx];
    float scale = wsv * wscale2[0];
    const float4* wp = (const float4*)(w + (size_t)idx * 16);
    Frag32B f;
#pragma unroll
    for (int q = 0; q < 4; ++q) {
        float4 v = wp[q];
        f.h[q * 4 + 0] = (__bf16)(e2m1_qdq(v.x / scale) * wsv);
        f.h[q * 4 + 1] = (__bf16)(e2m1_qdq(v.y / scale) * wsv);
        f.h[q * 4 + 2] = (__bf16)(e2m1_qdq(v.z / scale) * wsv);
        f.h[q * 4 + 3] = (__bf16)(e2m1_qdq(v.w / scale) * wsv);
    }
    uint4* op = (uint4*)(Bq + (size_t)idx * 16);
    op[0] = f.u[0];
    op[1] = f.u[1];
}

// ---------------------------------------------------------------------------
// Kernel 3: input QDQ -> bf16  Aq[m,k] = e2m1(x/(qs*s2)) * qs   (exact bf16)
// qs = fp8_round(pb==0 ? 1 : (amax_blk/6)/s2),  s2 = amax/6/448.
// ---------------------------------------------------------------------------
__global__ void __launch_bounds__(256)
xqdq_kernel(const float* __restrict__ x, const unsigned* __restrict__ amax_bits,
            __bf16* __restrict__ Aq, int M, int K) {
    int nkb = K >> 4;
    int idx = blockIdx.x * 256 + threadIdx.x;       // (m, kb) flat
    if (idx >= M * nkb) return;
    float amax = __uint_as_float(*amax_bits);
    float s2 = amax / 6.0f / 448.0f;

    const float4* xp = (const float4*)(x + (size_t)idx * 16);
    float4 v[4];
    float bmax = 0.0f;
#pragma unroll
    for (int q = 0; q < 4; ++q) {
        v[q] = xp[q];
        bmax = fmaxf(bmax, fmaxf(fmaxf(__builtin_fabsf(v[q].x), __builtin_fabsf(v[q].y)),
                                 fmaxf(__builtin_fabsf(v[q].z), __builtin_fabsf(v[q].w))));
    }
    float pb = bmax / 6.0f;
    float qs = (pb == 0.0f) ? 1.0f : pb / s2;
    qs = fp8_e4m3_round(qs);
    float denom = qs * s2;

    Frag32B f;
#pragma unroll
    for (int q = 0; q < 4; ++q) {
        f.h[q * 4 + 0] = (__bf16)(e2m1_qdq(v[q].x / denom) * qs);
        f.h[q * 4 + 1] = (__bf16)(e2m1_qdq(v[q].y / denom) * qs);
        f.h[q * 4 + 2] = (__bf16)(e2m1_qdq(v[q].z / denom) * qs);
        f.h[q * 4 + 3] = (__bf16)(e2m1_qdq(v[q].w / denom) * qs);
    }
    uint4* op = (uint4*)(Aq + (size_t)idx * 16);
    op[0] = f.u[0];
    op[1] = f.u[1];
}

// ---------------------------------------------------------------------------
// Kernel 4: WMMA GEMM  out[M,N] = g * Aq[M,K] @ Bq[N,K]^T + bias[N]
// 256 threads = 8 waves (2 along M x 4 along N). Block tile 128x256.
// Each wave: 64(M) x 64(N) = 4x4 accumulators of v_wmma_f32_16x16x32_bf16.
// Per 32-K step: 8 fragment loads (16 x b128) feed 16 WMMAs -> each fragment
// reused 4x in registers. A+B (96 MB bf16) reside in the 192 MB L2, so steady
// state loads are L2 hits; global_prefetch pulls the next K-slices into the
// WGP cache ahead of demand.
//
// Fragment layout per ISA (16-bit A 16x32): lane L holds row (L&15); K chunks
// at koff = (L>>4)*8 covering {koff..koff+7} and {koff+16..koff+23}. B uses
// the mirrored layout, so both sides are two 16-byte loads per lane from
// row-major [.,K] storage.
// ---------------------------------------------------------------------------
__device__ __forceinline__ void load_frag(Frag32B& f, const __bf16* p, int k0) {
    f.u[0] = *(const uint4*)(p + k0);
    f.u[1] = *(const uint4*)(p + k0 + 16);
}

__global__ void __launch_bounds__(256)
gemm_bf16_wmma_kernel(const __bf16* __restrict__ Aq, const __bf16* __restrict__ Bq,
                      const float* __restrict__ bias,
                      const unsigned* __restrict__ amax_bits,
                      const float* __restrict__ wscale2,
                      float* __restrict__ out, int M, int N, int K) {
    const int lane = threadIdx.x & 31;
    const int wave = threadIdx.x >> 5;
    const int wm = wave & 1;          // 0..1 -> 64-row slab
    const int wn = wave >> 1;         // 0..3 -> 64-col slab

    const int m0 = blockIdx.x * 128 + wm * 64;
    const int n0 = blockIdx.y * 256 + wn * 64;
    const int koff = (lane >> 4) << 3;

    const __bf16* pa[4];
    const __bf16* pb[4];
#pragma unroll
    for (int t = 0; t < 4; ++t) {
        pa[t] = Aq + (size_t)(m0 + t * 16 + (lane & 15)) * K + koff;
        pb[t] = Bq + (size_t)(n0 + t * 16 + (lane & 15)) * K + koff;
    }

    v8f acc[4][4];
#pragma unroll
    for (int i = 0; i < 4; ++i)
#pragma unroll
        for (int j = 0; j < 4; ++j) acc[i][j] = (v8f)(0.0f);

    const int PF = 256;               // prefetch distance: 8 K-steps (512 B)
    for (int k0 = 0; k0 < K; k0 += 32) {
        Frag32B a[4], b[4];
#pragma unroll
        for (int t = 0; t < 4; ++t) load_frag(a[t], pa[t], k0);
#pragma unroll
        for (int t = 0; t < 4; ++t) load_frag(b[t], pb[t], k0);

        if (k0 + PF < K) {
#pragma unroll
            for (int t = 0; t < 4; ++t) {
                __builtin_prefetch(pa[t] + k0 + PF, 0, 3);   // global_prefetch_b8
                __builtin_prefetch(pb[t] + k0 + PF, 0, 3);
            }
        }

#pragma unroll
        for (int i = 0; i < 4; ++i)
#pragma unroll
            for (int j = 0; j < 4; ++j)
                acc[i][j] = __builtin_amdgcn_wmma_f32_16x16x32_bf16(
                    false, a[i].v, false, b[j].v, (short)0, acc[i][j], false, false);
    }

    // epilogue: out = g * acc + bias;  g = s2 * ws2
    const float g = (__uint_as_float(*amax_bits) / 6.0f / 448.0f) * wscale2[0];
    // C layout: VGPR r -> M = r + 8*(lane>=16), N = lane&15
    const int mhalf = (lane >> 4) << 3;
    const int ncol = lane & 15;
#pragma unroll
    for (int i = 0; i < 4; ++i) {
        const int mbase = m0 + i * 16 + mhalf;
#pragma unroll
        for (int j = 0; j < 4; ++j) {
            const int n = n0 + j * 16 + ncol;
            const float bv = bias[n];
#pragma unroll
            for (int r = 0; r < 8; ++r) {
                out[(size_t)(mbase + r) * N + n] = g * acc[i][j][r] + bv;
            }
        }
    }
}

// ---------------------------------------------------------------------------
// Launch: inputs (x, weight, weight_scale, weight_scale_2, bias, block_size)
// ---------------------------------------------------------------------------
extern "C" void kernel_launch(void* const* d_in, const int* in_sizes, int n_in,
                              void* d_out, int out_size, void* d_ws, size_t ws_size,
                              hipStream_t stream) {
    const float* x       = (const float*)d_in[0];
    const float* weight  = (const float*)d_in[1];
    const float* wscale  = (const float*)d_in[2];
    const float* wscale2 = (const float*)d_in[3];
    const float* bias    = (const float*)d_in[4];

    const int O = in_sizes[4];                 // 4096
    const int K = in_sizes[1] / O;             // 4096
    const int M = in_sizes[0] / K;             // 8192 (B*S)
    (void)n_in; (void)out_size; (void)ws_size;

    // workspace layout
    char* wsb = (char*)d_ws;
    unsigned* amax = (unsigned*)wsb;                                 // 4 B
    __bf16* Aq = (__bf16*)(wsb + 256);                               // M*K*2
    __bf16* Bq = (__bf16*)(wsb + 256 + (size_t)M * K * 2);           // O*K*2

    hipMemsetAsync(amax, 0, sizeof(unsigned), stream);

    // 1) global amax(|x|)
    size_t n4 = (size_t)M * K / 4;
    amax_kernel<<<1024, 256, 0, stream>>>((const float4*)x, n4, amax);

    // 2) weight QDQ -> bf16
    int wblocks = (O * (K >> 4) + 255) / 256;
    wqdq_kernel<<<wblocks, 256, 0, stream>>>(weight, wscale, wscale2, Bq, O, K);

    // 3) input QDQ -> bf16 (needs amax)
    int xblocks = (M * (K >> 4) + 255) / 256;
    xqdq_kernel<<<xblocks, 256, 0, stream>>>(x, amax, Aq, M, K);

    // 4) WMMA GEMM + epilogue
    dim3 grid(M / 128, O / 256);
    gemm_bf16_wmma_kernel<<<grid, 256, 0, stream>>>(Aq, Bq, bias, amax, wscale2,
                                                    (float*)d_out, M, O, K);
}
